// SelfAttention_2095944040541
// MI455X (gfx1250) — compile-verified, matrix operand order checked by reading, not run on hardware
//
#include <hip/hip_runtime.h>
#include <math.h>

typedef __attribute__((ext_vector_type(16))) __bf16 v16bf;
typedef __attribute__((ext_vector_type(8)))  __bf16 v8bf;
typedef __attribute__((ext_vector_type(8)))  float  v8f;

#define B_    8
#define HW_   1024
#define C_    512
#define NH_   8
#define HD_   64
#define NG_   16
#define CG_   32      // C_/NG_
#define EPSV  1e-5f

// ---------------------------------------------------------------------------
// Fragment loader: 16-bit A/B operand, 16x32 tile, from an LDS array stored as
// [row][k] with pitch PITCH (bf16 units, multiple of 8 => 16B-aligned rows).
// Lane l reads row (rowbase + (l&15)); K elements come as two contiguous
// 8-bf16 (16-byte) runs at k = kb + h*8 and k = kb + 16 + h*8, h = l>>4,
// matching the CDNA5 16-bit A-matrix VGPR layout.  -> two ds_load_b128.
// ---------------------------------------------------------------------------
template<int PITCH>
__device__ inline v16bf frag_ab(const __bf16* lds, int rowbase, int kb) {
  const int lane = threadIdx.x & 31;
  const __bf16* p = lds + (size_t)(rowbase + (lane & 15)) * PITCH + kb + ((lane >> 4) << 3);
  union { v16bf v; v8bf q[2]; } u;
  u.q[0] = *(const v8bf*)(p);
  u.q[1] = *(const v8bf*)(p + 16);
  return u.v;
}

__device__ inline v8f wmma_bf16(v16bf a, v16bf b, v8f c) {
  return __builtin_amdgcn_wmma_f32_16x16x32_bf16(false, a, false, b, (short)0, c, false, false);
}

// ---------------------------------------------------------------------------
// Kernel 1: GroupNorm statistics.  One block per (batch, group): reduce
// 1024*32 = 32768 floats -> mean, rstd.
// ---------------------------------------------------------------------------
__global__ __launch_bounds__(256) void gn_stats(const float* __restrict__ x,
                                                float* __restrict__ mean,
                                                float* __restrict__ rstd) {
  const int bg = blockIdx.x;            // 0..127
  const int b = bg >> 4, g = bg & 15;
  const float* base = x + (size_t)b * HW_ * C_ + g * CG_;
  float s = 0.f, ss = 0.f;
  for (int i = threadIdx.x; i < HW_ * CG_; i += 256) {
    const int pos = i >> 5, cc = i & 31;
    const float v = base[(size_t)pos * C_ + cc];
    s += v; ss += v * v;
  }
  __shared__ float shs[256], shq[256];
  shs[threadIdx.x] = s; shq[threadIdx.x] = ss;
  __syncthreads();
  for (int off = 128; off > 0; off >>= 1) {
    if (threadIdx.x < off) {
      shs[threadIdx.x] += shs[threadIdx.x + off];
      shq[threadIdx.x] += shq[threadIdx.x + off];
    }
    __syncthreads();
  }
  if (threadIdx.x == 0) {
    const float inv_n = 1.0f / (float)(HW_ * CG_);
    const float m = shs[0] * inv_n;
    const float var = shq[0] * inv_n - m * m;
    mean[bg] = m;
    rstd[bg] = rsqrtf(var + EPSV);
  }
}

// ---------------------------------------------------------------------------
// Kernel 2: fused GroupNorm-normalize + QKV projection.
// Y = xn @ w_qkv : M=8192 (b*hw), N=1536, K=512.  Block tile 128x128,
// K-step 64 (16 WMMA per barrier pair).  8 waves (2M x 4N), each wave
// 64x32 = 4x2 accum tiles of 16x16.  global_prefetch of next k-tile.
// Epilogue scatters bf16 into q (scaled by hd^-0.5), k, v buffers laid out
// as (b*head, pos, hd).
// ---------------------------------------------------------------------------
#define P2 72   // LDS pitch in bf16 (144B: 16B aligned, 36-bank rotation)

__global__ __launch_bounds__(256) void qkv_gemm(
    const float* __restrict__ x, const float* __restrict__ gscale,
    const float* __restrict__ gbias, const float* __restrict__ w,
    const float* __restrict__ mean, const float* __restrict__ rstd,
    __bf16* __restrict__ qb, __bf16* __restrict__ kb, __bf16* __restrict__ vb) {
  __shared__ __align__(16) __bf16 la[128 * P2];   // [m][k]  normalized A tile
  __shared__ __align__(16) __bf16 lb[128 * P2];   // [n][k]  transposed B tile

  const int m0 = blockIdx.x * 128;
  const int n0 = blockIdx.y * 128;
  const int tid = threadIdx.x;
  const int wv = tid >> 5, lane = tid & 31;
  const int l15 = lane & 15, lh = lane >> 4;
  const int wm = (wv >> 2) * 64;        // wave M offset in tile
  const int wn = (wv & 3) * 32;         // wave N offset in tile
  const int b = m0 >> 10;               // whole tile in one batch (128 | 1024)

  v8f acc[4][2] = {};

  for (int k0 = 0; k0 < C_; k0 += 64) {
    // Two 32-channel groups covered by this 64-wide K tile.
    const int g0 = k0 >> 5;
    const float mu0 = mean[b * NG_ + g0],     rs0 = rstd[b * NG_ + g0];
    const float mu1 = mean[b * NG_ + g0 + 1], rs1 = rstd[b * NG_ + g0 + 1];
    // Stage A: 128 rows x 64 ch, normalize on the fly, fp32 -> bf16.
    for (int e = tid; e < 128 * 64; e += 256) {
      const int r = e >> 6, c = e & 63, ch = k0 + c;
      const float mu = (c < 32) ? mu0 : mu1;
      const float rs = (c < 32) ? rs0 : rs1;
      const float v = x[(size_t)(m0 + r) * C_ + ch];
      la[r * P2 + c] = (__bf16)((v - mu) * rs * gscale[ch] + gbias[ch]);
    }
    // Stage B transposed: lb[n][k] <- w[k0+k][n0+n]  (coalesced in n).
    for (int e = tid; e < 128 * 64; e += 256) {
      const int kk = e >> 7, nn = e & 127;
      lb[nn * P2 + kk] = (__bf16)w[(size_t)(k0 + kk) * (3 * C_) + n0 + nn];
    }
    // Prefetch next k-tile into L2 (global_prefetch_b8) while we compute:
    // x next tile = 128 rows x 256B (2 lines/row); w next = 64 rows x 4 lines.
    if (k0 + 64 < C_) {
      __builtin_prefetch(&x[(size_t)(m0 + (tid >> 1)) * C_ + k0 + 64 + (tid & 1) * 32], 0, 1);
      __builtin_prefetch(&w[(size_t)(k0 + 64 + (tid >> 2)) * (3 * C_) + n0 + (tid & 3) * 32], 0, 1);
    }
    __syncthreads();
#pragma unroll
    for (int kc = 0; kc < 64; kc += 32) {
      v16bf b0 = frag_ab<P2>(lb, wn,      kc);
      v16bf b1 = frag_ab<P2>(lb, wn + 16, kc);
#pragma unroll
      for (int i = 0; i < 4; ++i) {
        v16bf a = frag_ab<P2>(la, wm + i * 16, kc);
        acc[i][0] = wmma_bf16(a, b0, acc[i][0]);
        acc[i][1] = wmma_bf16(a, b1, acc[i][1]);
      }
    }
    __syncthreads();
  }

  // Epilogue: scatter to q/k/v (b*head, pos, hd) as bf16; q gets hd^-0.5.
#pragma unroll
  for (int i = 0; i < 4; ++i) {
#pragma unroll
    for (int j = 0; j < 2; ++j) {
      const int ncol = n0 + wn + j * 16 + l15;
      const int head = ncol / 192;
      const int rem  = ncol % 192;
      const int which = rem >> 6;           // 0=q 1=k 2=v
      const int ch = rem & 63;
      __bf16* dst = (which == 0) ? qb : ((which == 1) ? kb : vb);
      const float sc = (which == 0) ? 0.125f : 1.0f;   // hd^-0.5 = 1/8
#pragma unroll
      for (int r = 0; r < 8; ++r) {
        const int mrow = m0 + wm + i * 16 + lh * 8 + r;
        const int bb = mrow >> 10, pos = mrow & 1023;
        dst[(((size_t)(bb * NH_ + head)) * HW_ + pos) * HD_ + ch] =
            (__bf16)(acc[i][j][r] * sc);
      }
    }
  }
}

// ---------------------------------------------------------------------------
// Kernel 3: flash attention.  Block = (b*head, 128-query tile), 8 waves of
// 16 query rows each.  K-tiles of 64 keys; online softmax; P converted
// D-layout -> A-layout via per-wave LDS; O accumulated in fp32, written bf16
// directly in (b, pos, n*hd) layout for the out-projection.
// ---------------------------------------------------------------------------
#define AP 72   // LDS pitch in bf16 (144B: 16B aligned, 36-bank rotation)

__global__ __launch_bounds__(256) void attn(
    const __bf16* __restrict__ qb, const __bf16* __restrict__ kb,
    const __bf16* __restrict__ vb, __bf16* __restrict__ ob) {
  __shared__ __align__(16) __bf16 lq[128 * AP];        // [q][ch]
  __shared__ __align__(16) __bf16 lk[64 * AP];         // [kpos][ch]
  __shared__ __align__(16) __bf16 lvt[64 * AP];        // [ch][kpos]  (V^T)
  __shared__ __align__(16) __bf16 lp[8 * 16 * AP];     // per-wave P [16][64]

  const int bh = blockIdx.x;            // 0..63  (b*8 + head)
  const int q0 = blockIdx.y * 128;
  const int tid = threadIdx.x;
  const int wv = tid >> 5, lane = tid & 31;
  const int l15 = lane & 15, lh = lane >> 4;

  const __bf16* qsrc = qb + (size_t)bh * HW_ * HD_;
  const __bf16* ksrc = kb + (size_t)bh * HW_ * HD_;
  const __bf16* vsrc = vb + (size_t)bh * HW_ * HD_;

  // Load Q tile once (16-byte chunks).
  for (int e8 = tid; e8 < (128 * HD_) / 8; e8 += 256) {
    const int r = e8 >> 3, c8 = (e8 & 7) * 8;
    *(v8bf*)&lq[r * AP + c8] = *(const v8bf*)&qsrc[(size_t)(q0 + r) * HD_ + c8];
  }

  float mrow[8], lrow[8];
#pragma unroll
  for (int r = 0; r < 8; ++r) { mrow[r] = -1e30f; lrow[r] = 0.f; }
  v8f oacc[4] = {};

  for (int kt = 0; kt < HW_ / 64; ++kt) {
    const int kb0 = kt * 64;
    __syncthreads();
    // Stage K tile row-major.
    for (int e8 = tid; e8 < (64 * HD_) / 8; e8 += 256) {
      const int r = e8 >> 3, c8 = (e8 & 7) * 8;
      *(v8bf*)&lk[r * AP + c8] = *(const v8bf*)&ksrc[(size_t)(kb0 + r) * HD_ + c8];
    }
    // Stage V transposed: lvt[ch][kpos].
    for (int e = tid; e < 64 * HD_; e += 256) {
      const int kp = e >> 6, ch = e & 63;
      lvt[ch * AP + kp] = vsrc[(size_t)(kb0 + kp) * HD_ + ch];
    }
    // Prefetch next K/V tiles (64 rows x 128B = 1 cacheline per row each).
    if (kb0 + 64 < HW_) {
      if (tid < 64)        __builtin_prefetch(&ksrc[(size_t)(kb0 + 64 + tid) * HD_], 0, 1);
      else if (tid < 128)  __builtin_prefetch(&vsrc[(size_t)(kb0 + tid) * HD_], 0, 1);
    }
    __syncthreads();

    // S = Q * K^T  (16 q-rows x 64 k-cols per wave).
    v16bf aq0 = frag_ab<AP>(lq, wv * 16, 0);
    v16bf aq1 = frag_ab<AP>(lq, wv * 16, 32);
    v8f s[4] = {};
#pragma unroll
    for (int j = 0; j < 4; ++j) {
      s[j] = wmma_bf16(aq0, frag_ab<AP>(lk, j * 16, 0),  s[j]);
      s[j] = wmma_bf16(aq1, frag_ab<AP>(lk, j * 16, 32), s[j]);
    }

    // Online softmax: row reductions across the 16-lane halves (D layout:
    // N = lane&15, M = vgpr + 8*(lane>>4), so xor masks 1..8 stay in-row).
    float alpha[8];
#pragma unroll
    for (int r = 0; r < 8; ++r) {
      float v = fmaxf(fmaxf(s[0][r], s[1][r]), fmaxf(s[2][r], s[3][r]));
#pragma unroll
      for (int off = 1; off < 16; off <<= 1) v = fmaxf(v, __shfl_xor(v, off, 32));
      const float mn = fmaxf(mrow[r], v);
      alpha[r] = __expf(mrow[r] - mn);
      mrow[r] = mn;
    }
    float rsum[8];
#pragma unroll
    for (int r = 0; r < 8; ++r) rsum[r] = 0.f;
#pragma unroll
    for (int j = 0; j < 4; ++j)
#pragma unroll
      for (int r = 0; r < 8; ++r) {
        const float p = __expf(s[j][r] - mrow[r]);
        rsum[r] += p;
        lp[(size_t)(wv * 16 + lh * 8 + r) * AP + j * 16 + l15] = (__bf16)p;
      }
#pragma unroll
    for (int r = 0; r < 8; ++r) {
      float v = rsum[r];
#pragma unroll
      for (int off = 1; off < 16; off <<= 1) v += __shfl_xor(v, off, 32);
      lrow[r] = lrow[r] * alpha[r] + v;
    }
#pragma unroll
    for (int j = 0; j < 4; ++j)
#pragma unroll
      for (int r = 0; r < 8; ++r) oacc[j][r] *= alpha[r];

    // O += P * V   (per-wave LDS is private; DS ops are in-order per wave).
    const __bf16* lpw = lp + (size_t)wv * 16 * AP;
    v16bf ap0 = frag_ab<AP>(lpw, 0, 0);
    v16bf ap1 = frag_ab<AP>(lpw, 0, 32);
#pragma unroll
    for (int j = 0; j < 4; ++j) {
      oacc[j] = wmma_bf16(ap0, frag_ab<AP>(lvt, j * 16, 0),  oacc[j]);
      oacc[j] = wmma_bf16(ap1, frag_ab<AP>(lvt, j * 16, 32), oacc[j]);
    }
  }

  // Epilogue: O /= l, write bf16 into (b, pos, n*hd).
  const int bb = bh >> 3, head = bh & 7;
#pragma unroll
  for (int j = 0; j < 4; ++j)
#pragma unroll
    for (int r = 0; r < 8; ++r) {
      const int qpos = q0 + wv * 16 + lh * 8 + r;
      const int ch = j * 16 + l15;
      ob[((size_t)(bb * HW_ + qpos)) * C_ + head * HD_ + ch] =
          (__bf16)(oacc[j][r] / lrow[r]);
    }
}

// ---------------------------------------------------------------------------
// Kernel 4: output projection + bias + residual.
// out = o @ w_out + b_out + x : M=8192, N=512, K=512.  Same tiling as qkv
// (K-step 64, prefetch next k-tile).
// ---------------------------------------------------------------------------
__global__ __launch_bounds__(256) void out_gemm(
    const __bf16* __restrict__ o, const float* __restrict__ w,
    const float* __restrict__ bias, const float* __restrict__ xin,
    float* __restrict__ out) {
  __shared__ __align__(16) __bf16 la[128 * P2];
  __shared__ __align__(16) __bf16 lb[128 * P2];

  const int m0 = blockIdx.x * 128;
  const int n0 = blockIdx.y * 128;
  const int tid = threadIdx.x;
  const int wv = tid >> 5, lane = tid & 31;
  const int l15 = lane & 15, lh = lane >> 4;
  const int wm = (wv >> 2) * 64;
  const int wn = (wv & 3) * 32;

  v8f acc[4][2] = {};

  for (int k0 = 0; k0 < C_; k0 += 64) {
    // Stage A (already bf16): 16-byte chunked copy, 128 rows x 64 ch.
    for (int e8 = tid; e8 < (128 * 64) / 8; e8 += 256) {
      const int r = e8 >> 3, c8 = (e8 & 7) * 8;
      *(v8bf*)&la[r * P2 + c8] = *(const v8bf*)&o[(size_t)(m0 + r) * C_ + k0 + c8];
    }
    // Stage B transposed.
    for (int e = tid; e < 128 * 64; e += 256) {
      const int kk = e >> 7, nn = e & 127;
      lb[nn * P2 + kk] = (__bf16)w[(size_t)(k0 + kk) * C_ + n0 + nn];
    }
    // Prefetch next k-tile: o next = 128 rows x 1 line; w next = 64 x 4 lines.
    if (k0 + 64 < C_) {
      if (tid < 128) __builtin_prefetch(&o[(size_t)(m0 + tid) * C_ + k0 + 64], 0, 1);
      __builtin_prefetch(&w[(size_t)(k0 + 64 + (tid >> 2)) * C_ + n0 + (tid & 3) * 32], 0, 1);
    }
    __syncthreads();
#pragma unroll
    for (int kc = 0; kc < 64; kc += 32) {
      v16bf b0 = frag_ab<P2>(lb, wn,      kc);
      v16bf b1 = frag_ab<P2>(lb, wn + 16, kc);
#pragma unroll
      for (int i = 0; i < 4; ++i) {
        v16bf a = frag_ab<P2>(la, wm + i * 16, kc);
        acc[i][0] = wmma_bf16(a, b0, acc[i][0]);
        acc[i][1] = wmma_bf16(a, b1, acc[i][1]);
      }
    }
    __syncthreads();
  }

#pragma unroll
  for (int i = 0; i < 4; ++i)
#pragma unroll
    for (int j = 0; j < 2; ++j) {
      const int ncol = n0 + wn + j * 16 + l15;
      const float bn = bias[ncol];
#pragma unroll
      for (int r = 0; r < 8; ++r) {
        const int mrow = m0 + wm + i * 16 + lh * 8 + r;
        const size_t idx = (size_t)mrow * C_ + ncol;
        out[idx] = acc[i][j][r] + bn + xin[idx];
      }
    }
}

// ---------------------------------------------------------------------------
extern "C" void kernel_launch(void* const* d_in, const int* in_sizes, int n_in,
                              void* d_out, int out_size, void* d_ws, size_t ws_size,
                              hipStream_t stream) {
  const float* x      = (const float*)d_in[0];
  const float* gscale = (const float*)d_in[1];
  const float* gbias  = (const float*)d_in[2];
  const float* wqkv   = (const float*)d_in[3];
  const float* wout   = (const float*)d_in[4];
  const float* bout   = (const float*)d_in[5];
  float* out = (float*)d_out;

  char* ws = (char*)d_ws;
  float* mean = (float*)ws;              // 128 floats
  float* rstd = mean + 128;              // 128 floats
  const size_t HEADBUF = (size_t)B_ * NH_ * HW_ * HD_;   // 4M elems
  __bf16* qb = (__bf16*)(ws + 1024);
  __bf16* kb = qb + HEADBUF;
  __bf16* vb = kb + HEADBUF;
  __bf16* ob = vb + HEADBUF;             // (b, pos, n*hd) = 8192x512

  gn_stats<<<128, 256, 0, stream>>>(x, mean, rstd);
  qkv_gemm<<<dim3(64, 12), 256, 0, stream>>>(x, gscale, gbias, wqkv, mean, rstd,
                                             qb, kb, vb);
  attn<<<dim3(64, 8), 256, 0, stream>>>(qb, kb, vb, ob);
  out_gemm<<<dim3(64, 4), 256, 0, stream>>>(ob, wout, bout, x, out);
}
